// SmallWindowAttention_9380208575259
// MI455X (gfx1250) — matrix operand order repaired and checked by reading, not compiled, not run on hardware
//
#include <hip/hip_runtime.h>
#include <hip/hip_bf16.h>

typedef __bf16 bf16;
typedef bf16  v16bf  __attribute__((ext_vector_type(16)));
typedef bf16  bf16x2 __attribute__((ext_vector_type(2)));
typedef float v8f    __attribute__((ext_vector_type(8)));

#define DEV __device__ __forceinline__

DEV v8f v8f_zero() {
  v8f z = {0.f, 0.f, 0.f, 0.f, 0.f, 0.f, 0.f, 0.f};
  return z;
}

// A-matrix (16x32 MxK, bf16) per CDNA5 ISA:
// lane m = L%16 holds row m; half = L/16 selects K sub-ranges.
// VGPR i<4 : K = 2i + 8*half ; VGPR i>=4 : K = 16 + 2(i-4) + 8*half.
DEV v16bf load_a_tile(const bf16* base, int stride) {
  int lane = threadIdx.x & 31;
  int m = lane & 15, half = lane >> 4;
  const bf16* row = base + (size_t)m * stride + 8 * half;
  v16bf a;
#pragma unroll
  for (int i = 0; i < 4; ++i) {
    bf16x2 lo = *(const bf16x2*)(row + 2 * i);        // K = 8h+2i, +1
    bf16x2 hi = *(const bf16x2*)(row + 16 + 2 * i);   // K = 16+8h+2i, +1
    a[2 * i]     = lo[0];
    a[2 * i + 1] = lo[1];
    a[8 + 2 * i]     = hi[0];
    a[8 + 2 * i + 1] = hi[1];
  }
  return a;
}

// B-matrix (32x16 KxN, bf16): lane n = L%16 holds column n.
// VGPR i<4 : K = 2i + 8*half ; VGPR i>=4 : K = 16 + 2(i-4) + 8*half (same as A).
// `base` points at element [n=0][k=0] of a row-major (N x K) source,
// i.e. B[k][n] = base[n*stride + k]  (weights are [out,in] so this IS W^T).
DEV v16bf load_b_tile(const bf16* base, int stride) {
  int lane = threadIdx.x & 31;
  int n = lane & 15, half = lane >> 4;
  const bf16* col = base + (size_t)n * stride + 8 * half;
  v16bf b;
#pragma unroll
  for (int i = 0; i < 4; ++i) {
    bf16x2 kk = *(const bf16x2*)(col + 2 * i);
    bf16x2 kh = *(const bf16x2*)(col + 16 + 2 * i);
    b[2 * i]     = kk[0];
    b[2 * i + 1] = kk[1];
    b[8 + 2 * i]     = kh[0];
    b[8 + 2 * i + 1] = kh[1];
  }
  return b;
}

DEV v8f wmma_bf16(v16bf a, v16bf b, v8f c) {
  // (neg_a, A, neg_b, B, c_mod, C, reuse_a, reuse_b)
  return __builtin_amdgcn_wmma_f32_16x16x32_bf16(false, a, false, b, (short)0, c,
                                                 false, false);
}

// ---------------------------------------------------------------- convert ---
__global__ __launch_bounds__(256) void cvt_f32_bf16(const float* __restrict__ src,
                                                    bf16* __restrict__ dst, int n) {
  int i = blockIdx.x * blockDim.x + threadIdx.x;
  if (i < n) dst[i] = (bf16)src[i];
}

// --------------------------------------------------------------- qkv GEMM ---
// Y = x(8192x768) @ qkv_w^T(768x1536) + b ; scatter Q*SCALE -> (B,H,S,Hd),
// V -> transposed (B,H,Hd,S) so PV B-tiles are K-contiguous.
// WG tile 128x128: 4 M-waves x 2 N-waves; each wave 32x64 (2x4 WMMA tiles).
__global__ __launch_bounds__(256) void qkv_gemm_kernel(
    const bf16* __restrict__ xb, const bf16* __restrict__ wb,
    const float* __restrict__ bias, bf16* __restrict__ qb, bf16* __restrict__ vT) {
  const int K = 768;
  int wid = threadIdx.x >> 5, lane = threadIdx.x & 31;
  int mw = wid & 3, nw = wid >> 2;
  int rbase = blockIdx.x * 128 + mw * 32;
  int cbase = blockIdx.y * 128 + nw * 64;
  int lm = lane & 15;

  v8f acc[2][4];
#pragma unroll
  for (int ms = 0; ms < 2; ++ms)
#pragma unroll
    for (int ns = 0; ns < 4; ++ns) acc[ms][ns] = v8f_zero();

  const bf16* arow0 = xb + (size_t)(rbase + lm) * K;         // per-lane A rows
  const bf16* brow0 = wb + (size_t)(cbase + lm) * K;         // per-lane B rows

  for (int kk = 0; kk < K; kk += 32) {
    // prefetch next K-step's line segments (global_prefetch_b8)
    if (kk + 32 < K) {
      __builtin_prefetch(arow0 + kk + 32, 0, 1);
      __builtin_prefetch(arow0 + 16 * (size_t)K + kk + 32, 0, 1);
      __builtin_prefetch(brow0 + kk + 32, 0, 1);
      __builtin_prefetch(brow0 + 32 * (size_t)K + kk + 32, 0, 1);
    }
    v16bf a0 = load_a_tile(xb + (size_t)rbase * K + kk, K);
    v16bf a1 = load_a_tile(xb + (size_t)(rbase + 16) * K + kk, K);
    v16bf b0 = load_b_tile(wb + (size_t)cbase * K + kk, K);
    v16bf b1 = load_b_tile(wb + (size_t)(cbase + 16) * K + kk, K);
    v16bf b2 = load_b_tile(wb + (size_t)(cbase + 32) * K + kk, K);
    v16bf b3 = load_b_tile(wb + (size_t)(cbase + 48) * K + kk, K);
    acc[0][0] = wmma_bf16(a0, b0, acc[0][0]);
    acc[0][1] = wmma_bf16(a0, b1, acc[0][1]);
    acc[0][2] = wmma_bf16(a0, b2, acc[0][2]);
    acc[0][3] = wmma_bf16(a0, b3, acc[0][3]);
    acc[1][0] = wmma_bf16(a1, b0, acc[1][0]);
    acc[1][1] = wmma_bf16(a1, b1, acc[1][1]);
    acc[1][2] = wmma_bf16(a1, b2, acc[1][2]);
    acc[1][3] = wmma_bf16(a1, b3, acc[1][3]);
  }

  int half = lane >> 4;
#pragma unroll
  for (int ms = 0; ms < 2; ++ms)
#pragma unroll
    for (int ns = 0; ns < 4; ++ns) {
      int gn = cbase + ns * 16 + lm;
      float bv = bias[gn];
#pragma unroll
      for (int r = 0; r < 8; ++r) {
        int gm = rbase + ms * 16 + r + 8 * half;   // row in [0, 8192)
        float y = acc[ms][ns][r] + bv;
        int batch = gm >> 11, s = gm & 2047;
        if (gn < 768) {  // Q, pre-scaled by HEAD_DIM^-0.5 = 0.125
          int h = gn >> 6, d = gn & 63;
          qb[(((size_t)(batch * 12 + h)) * 2048 + s) * 64 + d] = (bf16)(y * 0.125f);
        } else {         // V, stored transposed (B,H,Hd,S)
          int e = gn - 768;
          int h = e >> 6, d = e & 63;
          vT[(((size_t)(batch * 12 + h)) * 64 + d) * 2048 + s] = (bf16)y;
        }
      }
    }
}

// -------------------------------------------------------- flash attention ---
// softmax(Qs Qs^T) V with online softmax; one wave owns 16 query rows.
__global__ __launch_bounds__(256) void attn_kernel(const bf16* __restrict__ qb,
                                                   const bf16* __restrict__ vT,
                                                   bf16* __restrict__ ao) {
  __shared__ bf16 Pst[8][16 * 32];   // per-wave P staging (C-layout -> A-layout)
  const int S = 2048, Hd = 64, H = 12, D = 768;
  int wid = threadIdx.x >> 5, lane = threadIdx.x & 31;
  int m = lane & 15, half = lane >> 4;

  int gw = blockIdx.x * 8 + wid;     // 6144 wave tasks
  int bh = gw >> 7, it = gw & 127;
  int b = bh / 12, h = bh % 12;
  const bf16* Q = qb + (size_t)(b * H + h) * S * Hd;   // (S, 64)
  const bf16* V = vT + (size_t)(b * H + h) * Hd * S;   // (64, S)
  int i0 = it * 16;

  v16bf aq0 = load_a_tile(Q + (size_t)i0 * Hd, Hd);       // K = 0..31
  v16bf aq1 = load_a_tile(Q + (size_t)i0 * Hd + 32, Hd);  // K = 32..63

  float mrow[8], lrow[8];
  v8f acc[4];
#pragma unroll
  for (int r = 0; r < 8; ++r) { mrow[r] = -3.0e38f; lrow[r] = 0.f; }
#pragma unroll
  for (int n = 0; n < 4; ++n) acc[n] = v8f_zero();

  bf16* myP = &Pst[wid][0];

  for (int j = 0; j < S; j += 32) {
    // prefetch next j-block operands while this one computes
    if (j + 32 < S) {
      __builtin_prefetch(Q + (size_t)(j + 32 + m) * Hd, 0, 1);
      __builtin_prefetch(Q + (size_t)(j + 48 + m) * Hd, 0, 1);
      __builtin_prefetch(V + (size_t)(m + 16 * half) * S + j + 32, 0, 1);
      __builtin_prefetch(V + (size_t)(m + 16 * half + 32) * S + j + 32, 0, 1);
    }
    // scores S_ij = Q_i (16x64) @ Q_j^T : Q rows j..j+31 act as B-tiles
    v16bf bl0 = load_b_tile(Q + (size_t)j * Hd, Hd);
    v16bf bl1 = load_b_tile(Q + (size_t)j * Hd + 32, Hd);
    v16bf br0 = load_b_tile(Q + (size_t)(j + 16) * Hd, Hd);
    v16bf br1 = load_b_tile(Q + (size_t)(j + 16) * Hd + 32, Hd);
    v8f sl = v8f_zero(), sr = v8f_zero();
    sl = wmma_bf16(aq0, bl0, sl);
    sl = wmma_bf16(aq1, bl1, sl);
    sr = wmma_bf16(aq0, br0, sr);
    sr = wmma_bf16(aq1, br1, sr);

    // online softmax stats: row r+8*half lives across the 16 lanes of `half`
    float alpha[8];
#pragma unroll
    for (int r = 0; r < 8; ++r) {
      float t = fmaxf(sl[r], sr[r]);
#pragma unroll
      for (int msk = 8; msk >= 1; msk >>= 1)
        t = fmaxf(t, __shfl_xor(t, msk, 32));
      float mn = fmaxf(mrow[r], t);
      alpha[r] = __expf(mrow[r] - mn);
      mrow[r] = mn;
      float pl = __expf(sl[r] - mn);
      float pr = __expf(sr[r] - mn);
      myP[(r + 8 * half) * 32 + m]      = (bf16)pl;   // cols j..j+15
      myP[(r + 8 * half) * 32 + 16 + m] = (bf16)pr;   // cols j+16..j+31
      float sum2 = pl + pr;
#pragma unroll
      for (int msk = 8; msk >= 1; msk >>= 1)
        sum2 += __shfl_xor(sum2, msk, 32);
      lrow[r] = lrow[r] * alpha[r] + sum2;
    }
#pragma unroll
    for (int n = 0; n < 4; ++n)
#pragma unroll
      for (int r = 0; r < 8; ++r) acc[n][r] *= alpha[r];

    // P (16x32) back out of LDS in A-layout; same-wave DS ops are in-order
    v16bf pa = load_a_tile(myP, 32);
#pragma unroll
    for (int n = 0; n < 4; ++n) {
      v16bf bv = load_b_tile(V + (size_t)(n * 16) * S + j, S);  // K-contiguous
      acc[n] = wmma_bf16(pa, bv, acc[n]);
    }
  }

#pragma unroll
  for (int r = 0; r < 8; ++r) {
    float inv = 1.0f / lrow[r];
    int s = i0 + r + 8 * half;
#pragma unroll
    for (int n = 0; n < 4; ++n) {
      int dcol = h * 64 + n * 16 + m;
      ao[(size_t)(b * S + s) * D + dcol] = (bf16)(acc[n][r] * inv);
    }
  }
}

// -------------------------------------------------------------- proj GEMM ---
// WG tile 128x128, wave tile 32x64, same scheme as qkv GEMM; f32 output.
__global__ __launch_bounds__(256) void proj_gemm_kernel(
    const bf16* __restrict__ ab, const bf16* __restrict__ wb,
    const float* __restrict__ bias, float* __restrict__ out) {
  const int K = 768;
  int wid = threadIdx.x >> 5, lane = threadIdx.x & 31;
  int mw = wid & 3, nw = wid >> 2;
  int rbase = blockIdx.x * 128 + mw * 32;
  int cbase = blockIdx.y * 128 + nw * 64;
  int lm = lane & 15;

  v8f acc[2][4];
#pragma unroll
  for (int ms = 0; ms < 2; ++ms)
#pragma unroll
    for (int ns = 0; ns < 4; ++ns) acc[ms][ns] = v8f_zero();

  const bf16* arow0 = ab + (size_t)(rbase + lm) * K;
  const bf16* brow0 = wb + (size_t)(cbase + lm) * K;

  for (int kk = 0; kk < K; kk += 32) {
    if (kk + 32 < K) {
      __builtin_prefetch(arow0 + kk + 32, 0, 1);
      __builtin_prefetch(arow0 + 16 * (size_t)K + kk + 32, 0, 1);
      __builtin_prefetch(brow0 + kk + 32, 0, 1);
      __builtin_prefetch(brow0 + 32 * (size_t)K + kk + 32, 0, 1);
    }
    v16bf a0 = load_a_tile(ab + (size_t)rbase * K + kk, K);
    v16bf a1 = load_a_tile(ab + (size_t)(rbase + 16) * K + kk, K);
    v16bf b0 = load_b_tile(wb + (size_t)cbase * K + kk, K);
    v16bf b1 = load_b_tile(wb + (size_t)(cbase + 16) * K + kk, K);
    v16bf b2 = load_b_tile(wb + (size_t)(cbase + 32) * K + kk, K);
    v16bf b3 = load_b_tile(wb + (size_t)(cbase + 48) * K + kk, K);
    acc[0][0] = wmma_bf16(a0, b0, acc[0][0]);
    acc[0][1] = wmma_bf16(a0, b1, acc[0][1]);
    acc[0][2] = wmma_bf16(a0, b2, acc[0][2]);
    acc[0][3] = wmma_bf16(a0, b3, acc[0][3]);
    acc[1][0] = wmma_bf16(a1, b0, acc[1][0]);
    acc[1][1] = wmma_bf16(a1, b1, acc[1][1]);
    acc[1][2] = wmma_bf16(a1, b2, acc[1][2]);
    acc[1][3] = wmma_bf16(a1, b3, acc[1][3]);
  }

  int half = lane >> 4;
#pragma unroll
  for (int ms = 0; ms < 2; ++ms)
#pragma unroll
    for (int ns = 0; ns < 4; ++ns) {
      int gn = cbase + ns * 16 + lm;
      float bv = bias[gn];
#pragma unroll
      for (int r = 0; r < 8; ++r) {
        int gm = rbase + ms * 16 + r + 8 * half;
        out[(size_t)gm * 768 + gn] = acc[ms][ns][r] + bv;
      }
    }
}

// ------------------------------------------------------------------ launch ---
extern "C" void kernel_launch(void* const* d_in, const int* in_sizes, int n_in,
                              void* d_out, int out_size, void* d_ws, size_t ws_size,
                              hipStream_t stream) {
  const float* x      = (const float*)d_in[0];
  const float* qkv_w  = (const float*)d_in[1];
  const float* qkv_b  = (const float*)d_in[2];
  const float* proj_w = (const float*)d_in[3];
  const float* proj_b = (const float*)d_in[4];
  float* out = (float*)d_out;

  const size_t N_X  = 8192ull * 768;        // x (and attn-out alias)
  const size_t N_WQ = 1536ull * 768;
  const size_t N_WP = 768ull * 768;
  const size_t N_QV = 4ull * 12 * 2048 * 64;

  char* p = (char*)d_ws;
  auto carve = [&](size_t elems) {
    bf16* r = (bf16*)p;
    p += ((elems * sizeof(bf16) + 255) & ~(size_t)255);
    return r;
  };
  bf16* xb = carve(N_X);
  bf16* wq = carve(N_WQ);
  bf16* wp = carve(N_WP);
  bf16* qb = carve(N_QV);
  bf16* vT = carve(N_QV);
  bf16* ao = xb;  // x no longer needed once attention runs

  cvt_f32_bf16<<<(int)((N_X + 255) / 256), 256, 0, stream>>>(x, xb, (int)N_X);
  cvt_f32_bf16<<<(int)((N_WQ + 255) / 256), 256, 0, stream>>>(qkv_w, wq, (int)N_WQ);
  cvt_f32_bf16<<<(int)((N_WP + 255) / 256), 256, 0, stream>>>(proj_w, wp, (int)N_WP);

  qkv_gemm_kernel<<<dim3(64, 12), 256, 0, stream>>>(xb, wq, qkv_b, qb, vT);
  attn_kernel<<<768, 256, 0, stream>>>(qb, vT, ao);
  proj_gemm_kernel<<<dim3(64, 6), 256, 0, stream>>>(ao, wp, proj_b, out);
}
